// MultiHeadAttention_68178310857181
// MI455X (gfx1250) — compile-verified, matrix operand order checked by reading, not run on hardware
//
#include <hip/hip_runtime.h>

// ---------------------------------------------------------------------------
// Multi-head attention for MI455X (gfx1250, wave32, WMMA 16x16x32 f16,
// async global->LDS staging). B=4, S=2048, D=1024, H=16, Dh=64.
// ---------------------------------------------------------------------------

typedef _Float16 half_t;
typedef __attribute__((ext_vector_type(16))) _Float16 v16h;
typedef __attribute__((ext_vector_type(8)))  _Float16 v8h;
typedef __attribute__((ext_vector_type(8)))  float    v8f;

#define BATCH 4
#define SEQ   2048
#define DIM   1024
#define HEADS 16
#define DHEAD 64
#define MROWS (BATCH * SEQ)   // 8192

__device__ __forceinline__ v8f wmma32(v16h a, v16h b, v8f c) {
  // D = A(16x32) * B(32x16) + C(16x16), f16 in / f32 accumulate
  return __builtin_amdgcn_wmma_f32_16x16x32_f16(
      /*neg_a=*/false, a, /*neg_b=*/false, b,
      /*c_mod=*/(short)0, c, /*reuse_a=*/false, /*reuse_b=*/false);
}

// Load a 16x32 f16 fragment in the CDNA5 A/B VGPR striping.
// Lanes 0-15 hold K = {kb..kb+7, kb+16..kb+23}; lanes 16-31 hold
// K = {kb+8..kb+15, kb+24..kb+31}. Two contiguous 16-byte loads per lane.
__device__ __forceinline__ v16h load_frag16x32(const half_t* __restrict__ base,
                                               int ld, int row0, int kb) {
  const int lane = threadIdx.x & 31;
  const int hl   = lane >> 4;
  const int lm   = lane & 15;
  const half_t* p = base + (size_t)(row0 + lm) * ld + kb + hl * 8;
  v16h r;
  ((v8h*)&r)[0] = *(const v8h*)(p);
  ((v8h*)&r)[1] = *(const v8h*)(p + 16);
  return r;
}

// Async 16-byte copy global -> LDS (GLOBAL_LOAD_ASYNC_TO_LDS_B128,
// tracked on ASYNCcnt). ldx_byte is the raw LDS byte address: the LDS
// aperture maps generic addr[31:0] directly to the LDS offset (ISA 10.2).
__device__ __forceinline__ void async_copy16(unsigned lds_byte, const void* gptr) {
  asm volatile("global_load_async_to_lds_b128 %0, %1, off"
               :
               : "v"(lds_byte), "v"((unsigned long long)(size_t)gptr)
               : "memory");
}

__device__ __forceinline__ void wait_async0() {
  asm volatile("s_wait_asynccnt 0x0" ::: "memory");
}

// --------------------------- conversion kernels ----------------------------

__global__ void cvt_f32_to_f16_kernel(const float* __restrict__ in,
                                      half_t* __restrict__ out, int n) {
  int i = blockIdx.x * blockDim.x + threadIdx.x;
  if (i < n) out[i] = (half_t)in[i];
}

// W is [K][N] row-major f32; write Wt[N][K] f16 so WMMA B fragments load
// contiguously along the contraction dimension.
__global__ void cvt_transpose_kernel(const float* __restrict__ W,
                                     half_t* __restrict__ Wt, int K, int N) {
  int idx = blockIdx.x * blockDim.x + threadIdx.x;
  if (idx >= K * N) return;
  int n = idx / K;
  int k = idx - n * K;
  Wt[(size_t)n * K + k] = (half_t)W[(size_t)k * N + n];
}

// ----------------------------- projection GEMM -----------------------------
// Y[M,N] = A[M,K] (f16) * Bt[N,K]^T (f16) + bias, f32 accum via WMMA.
// 256 threads = 8 waves arranged 4(M) x 2(N); each wave register-blocks a
// 32x64 output (2x4 tiles of 16x16) -> 8 WMMAs per 12 b128 loads per K-step.
// Block tile 128x128.
// mode 0: f16 head-major [B,H,S,Dh] (Q,K); mode 1: f16 [B,H,Dh,S] (V^T);
// mode 2: f32 plain [M,N] (final output).
__global__ void proj_gemm_kernel(const half_t* __restrict__ A,
                                 const half_t* __restrict__ Bt,
                                 const float* __restrict__ bias,
                                 void* __restrict__ out,
                                 int M, int N, int K, int mode) {
  const int wid  = threadIdx.x >> 5;     // 0..7
  const int lane = threadIdx.x & 31;
  const int hl   = lane >> 4;
  const int lm   = lane & 15;
  const int wm   = wid & 3;              // 4 waves along M
  const int wn   = wid >> 2;             // 2 waves along N
  const int m0   = blockIdx.x * 128 + wm * 32;
  const int n0   = blockIdx.y * 128 + wn * 64;

  v8f acc[2][4] = {};
  for (int kb = 0; kb < K; kb += 32) {
    v16h a0 = load_frag16x32(A, K, m0,      kb);
    v16h a1 = load_frag16x32(A, K, m0 + 16, kb);
    v16h b0 = load_frag16x32(Bt, K, n0,      kb);
    v16h b1 = load_frag16x32(Bt, K, n0 + 16, kb);
    v16h b2 = load_frag16x32(Bt, K, n0 + 32, kb);
    v16h b3 = load_frag16x32(Bt, K, n0 + 48, kb);
    acc[0][0] = wmma32(a0, b0, acc[0][0]);
    acc[0][1] = wmma32(a0, b1, acc[0][1]);
    acc[0][2] = wmma32(a0, b2, acc[0][2]);
    acc[0][3] = wmma32(a0, b3, acc[0][3]);
    acc[1][0] = wmma32(a1, b0, acc[1][0]);
    acc[1][1] = wmma32(a1, b1, acc[1][1]);
    acc[1][2] = wmma32(a1, b2, acc[1][2]);
    acc[1][3] = wmma32(a1, b3, acc[1][3]);
  }

  for (int tn = 0; tn < 4; ++tn) {
    const int   n  = n0 + tn * 16 + lm;
    const float bn = bias ? bias[n] : 0.0f;
    for (int tm = 0; tm < 2; ++tm) {
      for (int r = 0; r < 8; ++r) {
        const int   m = m0 + tm * 16 + r + hl * 8;   // C layout
        const float v = acc[tm][tn][r] + bn;
        if (mode == 2) {
          ((float*)out)[(size_t)m * N + n] = v;
        } else {
          const int h  = n >> 6;
          const int d  = n & 63;
          const int b_ = m / SEQ;
          const int s  = m - b_ * SEQ;
          size_t off;
          if (mode == 0)
            off = (((size_t)b_ * HEADS + h) * SEQ + s) * DHEAD + d;
          else
            off = (((size_t)b_ * HEADS + h) * DHEAD + d) * (size_t)SEQ + s;
          ((half_t*)out)[off] = (half_t)v;
        }
      }
    }
  }
}

// ------------------------------- attention ---------------------------------
// 128 threads = 4 waves; each wave owns one 16-row query tile of one (b,h).
// All waves share per-key-step K (32x64) and V^T (64x32) tiles staged into
// LDS via async global->LDS copies. Flash-style online softmax in f32.
__global__ void attn_kernel(const half_t* __restrict__ Q,
                            const half_t* __restrict__ Km,
                            const half_t* __restrict__ Vt,
                            half_t* __restrict__ Out) {
  __shared__ __align__(16) half_t kbuf[32 * 64];        // keys j..j+31, [key][dh]
  __shared__ __align__(16) half_t vbuf[64 * 32];        // [dh][key-in-chunk]
  __shared__ __align__(16) half_t pshare[4][16 * 32];   // per-wave P tile

  const int tid  = threadIdx.x;          // 0..127
  const int wid  = tid >> 5;             // 0..3
  const int lane = tid & 31;
  const int hl   = lane >> 4;
  const int lm   = lane & 15;
  const int bh   = blockIdx.y;
  const int q0   = blockIdx.x * 64 + wid * 16;

  const half_t* Qp = Q  + (size_t)bh * SEQ * DHEAD;
  const half_t* Kp = Km + (size_t)bh * SEQ * DHEAD;
  const half_t* Vp = Vt + (size_t)bh * DHEAD * SEQ;

  const unsigned kbase = (unsigned)(size_t)(void*)kbuf;  // raw LDS byte addr
  const unsigned vbase = (unsigned)(size_t)(void*)vbuf;

  // Pre-scaled Q fragments (1/sqrt(64) = 0.125); Dh=64 -> two K-steps.
  v16h qf0 = load_frag16x32(Qp, DHEAD, q0, 0);
  v16h qf1 = load_frag16x32(Qp, DHEAD, q0, 32);
  const half_t sc = (half_t)0.125f;
  for (int i = 0; i < 16; ++i) { qf0[i] *= sc; qf1[i] *= sc; }

  v8f o0 = {}, o1 = {}, o2 = {}, o3 = {};
  float mrow[8], lrow[8];
  for (int r = 0; r < 8; ++r) { mrow[r] = -1e30f; lrow[r] = 0.0f; }

  half_t* ps = &pshare[wid][0];

  for (int j = 0; j < SEQ; j += 32) {
    // ---- async stage: K chunk is 32 rows x 64 dh = 4KB contiguous;
    //      V^T chunk is 64 rows (dh) x 32 keys, 64B per row = 4KB.
    {
      const half_t* gk = Kp + (size_t)j * DHEAD;  // contiguous 2048 halfs
      const int c0 = tid, c1 = tid + 128;         // 16B chunk ids, 256 total
      async_copy16(kbase + c0 * 16, gk + c0 * 8);
      async_copy16(kbase + c1 * 16, gk + c1 * 8);
      // vbuf element offset of chunk c is c*8 ( = (c>>2)*32 + (c&3)*8 )
      async_copy16(vbase + c0 * 16, Vp + (size_t)(c0 >> 2) * SEQ + j + (c0 & 3) * 8);
      async_copy16(vbase + c1 * 16, Vp + (size_t)(c1 >> 2) * SEQ + j + (c1 & 3) * 8);
    }
    wait_async0();
    __syncthreads();   // K/V tiles visible to all waves

    // ---- scores for keys [j, j+32): contraction over Dh=64 (kb 0 and 32)
    v16h k0a = load_frag16x32(kbuf, DHEAD, 0,  0);
    v16h k0b = load_frag16x32(kbuf, DHEAD, 0,  32);
    v16h k1a = load_frag16x32(kbuf, DHEAD, 16, 0);
    v16h k1b = load_frag16x32(kbuf, DHEAD, 16, 32);
    v8f s0 = {}, s1 = {};
    s0 = wmma32(qf0, k0a, s0);  s0 = wmma32(qf1, k0b, s0);
    s1 = wmma32(qf0, k1a, s1);  s1 = wmma32(qf1, k1b, s1);

    // ---- online softmax; rows = (r, lane-half), cols = lm across 16 lanes
    for (int r = 0; r < 8; ++r) {
      float mx = fmaxf(s0[r], s1[r]);
      for (int off = 1; off < 16; off <<= 1)
        mx = fmaxf(mx, __shfl_xor(mx, off, 32));
      const float mnew = fmaxf(mrow[r], mx);
      const float corr = __expf(mrow[r] - mnew);
      const float p0   = __expf(s0[r] - mnew);
      const float p1   = __expf(s1[r] - mnew);
      float sum = p0 + p1;
      for (int off = 1; off < 16; off <<= 1)
        sum += __shfl_xor(sum, off, 32);
      lrow[r] = lrow[r] * corr + sum;
      mrow[r] = mnew;
      o0[r] *= corr; o1[r] *= corr; o2[r] *= corr; o3[r] *= corr;
      const int row = r + hl * 8;
      ps[row * 32 + lm]      = (half_t)p0;
      ps[row * 32 + 16 + lm] = (half_t)p1;
    }
    __syncthreads();   // P writes visible before fragment re-load

    // ---- P (16x32, A-frag from LDS) x V^T tiles (B-frag from LDS)
    v16h pf = load_frag16x32(ps, 32, 0, 0);
    v16h v0 = load_frag16x32(vbuf, 32, 0,  0);
    v16h v1 = load_frag16x32(vbuf, 32, 16, 0);
    v16h v2 = load_frag16x32(vbuf, 32, 32, 0);
    v16h v3 = load_frag16x32(vbuf, 32, 48, 0);
    o0 = wmma32(pf, v0, o0);
    o1 = wmma32(pf, v1, o1);
    o2 = wmma32(pf, v2, o2);
    o3 = wmma32(pf, v3, o3);
    __syncthreads();   // all LDS reads done before next iter's async writes
  }

  // ---- normalize, store concat layout [B,S,D], D index = h*64 + d
  const int b_ = bh >> 4;
  const int h  = bh & 15;
  for (int r = 0; r < 8; ++r) {
    const float inv = 1.0f / lrow[r];
    const int   s   = q0 + r + hl * 8;
    const size_t base = ((size_t)b_ * SEQ + s) * DIM + h * DHEAD;
    Out[base +  0 + lm] = (half_t)(o0[r] * inv);
    Out[base + 16 + lm] = (half_t)(o1[r] * inv);
    Out[base + 32 + lm] = (half_t)(o2[r] * inv);
    Out[base + 48 + lm] = (half_t)(o3[r] * inv);
  }
}

// ------------------------------- launcher ----------------------------------

extern "C" void kernel_launch(void* const* d_in, const int* in_sizes, int n_in,
                              void* d_out, int out_size, void* d_ws, size_t ws_size,
                              hipStream_t stream) {
  (void)in_sizes; (void)n_in; (void)out_size; (void)ws_size;

  const float* X  = (const float*)d_in[0];
  const float* Wq = (const float*)d_in[1];
  const float* bq = (const float*)d_in[2];
  const float* Wk = (const float*)d_in[3];
  const float* bk = (const float*)d_in[4];
  const float* Wv = (const float*)d_in[5];
  const float* bv = (const float*)d_in[6];
  const float* Wo = (const float*)d_in[7];
  const float* bo = (const float*)d_in[8];

  char*  ws  = (char*)d_ws;
  size_t off = 0;
  auto walloc = [&](size_t bytes) -> void* {
    void* p = ws + off;
    off = (off + bytes + 255) & ~(size_t)255;
    return p;
  };

  const size_t XE = (size_t)MROWS * DIM;
  half_t* Xh  = (half_t*)walloc(XE * sizeof(half_t));
  half_t* Wqt = (half_t*)walloc((size_t)DIM * DIM * sizeof(half_t));
  half_t* Wkt = (half_t*)walloc((size_t)DIM * DIM * sizeof(half_t));
  half_t* Wvt = (half_t*)walloc((size_t)DIM * DIM * sizeof(half_t));
  half_t* Wot = (half_t*)walloc((size_t)DIM * DIM * sizeof(half_t));
  half_t* Qh  = (half_t*)walloc(XE * sizeof(half_t));   // [B,H,S,Dh]
  half_t* Kh  = (half_t*)walloc(XE * sizeof(half_t));   // [B,H,S,Dh]
  half_t* Vth = (half_t*)walloc(XE * sizeof(half_t));   // [B,H,Dh,S]
  half_t* Ch  = Xh;   // reuse: X no longer needed after QKV projections

  // 1) conversions
  cvt_f32_to_f16_kernel<<<(int)((XE + 255) / 256), 256, 0, stream>>>(X, Xh, (int)XE);
  const int WN = DIM * DIM;
  cvt_transpose_kernel<<<(WN + 255) / 256, 256, 0, stream>>>(Wq, Wqt, DIM, DIM);
  cvt_transpose_kernel<<<(WN + 255) / 256, 256, 0, stream>>>(Wk, Wkt, DIM, DIM);
  cvt_transpose_kernel<<<(WN + 255) / 256, 256, 0, stream>>>(Wv, Wvt, DIM, DIM);
  cvt_transpose_kernel<<<(WN + 255) / 256, 256, 0, stream>>>(Wo, Wot, DIM, DIM);

  // 2) QKV projections (register-blocked WMMA GEMM, block tile 128x128)
  dim3 gGemm(MROWS / 128, DIM / 128);   // (64, 8)
  proj_gemm_kernel<<<gGemm, 256, 0, stream>>>(Xh, Wqt, bq, Qh,  MROWS, DIM, DIM, 0);
  proj_gemm_kernel<<<gGemm, 256, 0, stream>>>(Xh, Wkt, bk, Kh,  MROWS, DIM, DIM, 0);
  proj_gemm_kernel<<<gGemm, 256, 0, stream>>>(Xh, Wvt, bv, Vth, MROWS, DIM, DIM, 1);

  // 3) attention (WMMA flash-style with async K/V staging into LDS)
  dim3 gAttn(SEQ / 64, BATCH * HEADS);  // (32, 64)
  attn_kernel<<<gAttn, 128, 0, stream>>>(Qh, Kh, Vth, Ch);

  // 4) output projection -> f32 d_out
  proj_gemm_kernel<<<gGemm, 256, 0, stream>>>(Ch, Wot, bo, d_out, MROWS, DIM, DIM, 2);
}
